// GATEncoder_47210280518123
// MI455X (gfx1250) — matrix operand order repaired and checked by reading, not compiled, not run on hardware
//
#include <hip/hip_runtime.h>
#include <math.h>

// ---------------------------------------------------------------------------
// GAT encoder pipeline for gfx1250 (MI455X).
// Dense GEMMs -> v_wmma_f32_16x16x32_bf16, K-unrolled, bf16-packed operands
//   (A: two b128 loads/lane/step, B: one 32B fragment load/lane/step).
// Edge softmax/aggregation -> L2-resident gather/atomic kernels
//   (all working buffers < 192 MB L2; HBM is never the bottleneck).
// ---------------------------------------------------------------------------

typedef __attribute__((ext_vector_type(16))) __bf16 v16bf;
typedef __attribute__((ext_vector_type(8)))  __bf16 v8bf;
typedef __attribute__((ext_vector_type(8)))  float  v8f;

static constexpr int kN     = 50000;   // nodes
static constexpr int kIn    = 128;     // input features
static constexpr int kHeads = 4;
static constexpr int kC1    = 64;
static constexpr int kTC1   = kHeads * kC1;  // 256
static constexpr int kC2    = 32;
static constexpr int kE     = 800000;  // 750000 random + 50000 self loops
static constexpr int kNB    = 4096;
static constexpr int kProj  = 32;
static constexpr float kSlope = 0.2f;

// ---- workspace layout (float element offsets; all sizes multiples of 64) ---
static constexpr size_t OFF_H1    = 0;                             // N*256
static constexpr size_t OFF_AGGR1 = OFF_H1    + (size_t)kN*kTC1;   // N*256
static constexpr size_t OFF_SSRC  = OFF_AGGR1 + (size_t)kN*kTC1;   // N*4
static constexpr size_t OFF_SDST  = OFF_SSRC  + (size_t)kN*kHeads;
static constexpr size_t OFF_MORD  = OFF_SDST  + (size_t)kN*kHeads; // uint
static constexpr size_t OFF_Z     = OFF_MORD  + (size_t)kN*kHeads;
static constexpr size_t OFF_PARR  = OFF_Z     + (size_t)kN*kHeads; // E*4
static constexpr size_t OFF_X2A   = OFF_PARR  + (size_t)kE*kHeads; // N*32
static constexpr size_t OFF_AGGR2 = OFF_X2A   + (size_t)kN*kC2;    // N*32
static constexpr size_t OFF_HDEC  = OFF_AGGR2 + (size_t)kN*kC2;    // 4096*32
static constexpr size_t OFF_WRS   = OFF_HDEC  + (size_t)kNB*kC2;   // 64
// bf16 regions (sizes below are in FLOAT units = bf16_count/2)
static constexpr size_t OFF_ABF   = OFF_WRS   + 64;                 // N*256 bf16
static constexpr size_t OFF_X2OBF = OFF_ABF   + (size_t)kN*kTC1/2;  // N*32 bf16
static constexpr size_t OFF_X2ABF = OFF_X2OBF + (size_t)kN*kC2/2;
static constexpr size_t OFF_CATBF = OFF_X2ABF + (size_t)kN*kC2/2;   // 4096*64 bf16
static constexpr size_t OFF_W1P   = OFF_CATBF + (size_t)kNB*2*kC2/2;
static constexpr size_t OFF_W2P   = OFF_W1P   + (size_t)kIn*kTC1/2;
static constexpr size_t OFF_WQP   = OFF_W2P   + (size_t)kTC1*kC2/2;
static constexpr size_t OFF_WKP   = OFF_WQP   + (size_t)kC2*kProj/2;
static constexpr size_t OFF_WFP   = OFF_WKP   + (size_t)kC2*kProj/2;

static inline int cdiv(long long a, long long b) { return (int)((a + b - 1) / b); }

// ---- helpers ---------------------------------------------------------------
__device__ __forceinline__ __bf16 f2bf(float f) {
  unsigned u = __builtin_bit_cast(unsigned, f);
  unsigned r = u + 0x7FFFu + ((u >> 16) & 1u);   // round-to-nearest-even
  unsigned short h = (unsigned short)(r >> 16);
  return __builtin_bit_cast(__bf16, h);
}
__device__ __forceinline__ unsigned f2ord(float f) {  // monotone float->uint
  unsigned u = __builtin_bit_cast(unsigned, f);
  return (u & 0x80000000u) ? ~u : (u | 0x80000000u);
}
__device__ __forceinline__ float ord2f(unsigned u) {
  unsigned v = (u & 0x80000000u) ? (u & 0x7FFFFFFFu) : ~u;
  return __builtin_bit_cast(float, v);
}
__device__ __forceinline__ float lrelu(float v) { return v > 0.f ? v : kSlope * v; }

// ---- f32 -> bf16 row-major convert -----------------------------------------
__global__ void conv_bf16_k(const float* __restrict__ in, __bf16* __restrict__ out,
                            long long n) {
  long long i = (long long)blockIdx.x * blockDim.x + threadIdx.x;
  if (i < n) out[i] = f2bf(in[i]);
}

// ---- pack B[K,Nc] (f32 row-major) into per-32K-block column-major bf16 -----
// Bp[((k/32)*Nc + n)*32 + (k%32)] : a lane's 16-element fragment is contiguous.
__global__ void pack_b_k(const float* __restrict__ B, __bf16* __restrict__ Bp,
                         int K, int Nc) {
  int t = blockIdx.x * blockDim.x + threadIdx.x;
  if (t >= K * Nc) return;
  int k = t / Nc, n = t % Nc;
  Bp[((size_t)(k >> 5) * Nc + n) * 32 + (k & 31)] = f2bf(B[t]);
}

// ---- bf16 WMMA GEMM: C[M,Nc] = A[M,K] @ B[K,Nc], K compile-time ------------
// One wave per 16x16 tile; fully unrolled K loop -> back-to-back v_wmma.
template <int K>
__global__ __launch_bounds__(256)
void wmma_gemm_t(const __bf16* __restrict__ A, const __bf16* __restrict__ Bp,
                 float* __restrict__ C, int M, int Nc) {
  const int lane = threadIdx.x & 31;
  const int wave = blockIdx.x * (blockDim.x >> 5) + (threadIdx.x >> 5);
  const int tilesN = Nc >> 4;
  const int nTiles = (M >> 4) * tilesN;
  if (wave >= nTiles) return;                    // whole-wave uniform exit
  const int tm = wave / tilesN, tn = wave % tilesN;
  const int half = lane >> 4, lq = lane & 15;

  const __bf16* arow = A + (size_t)(tm * 16 + lq) * K;  // A row for this lane
  const int ncol = tn * 16 + lq;                        // B column for this lane
  v8f acc = {};
#pragma unroll
  for (int k0 = 0; k0 < K; k0 += 32) {
    // A 16x32 fragment: element i -> K=k0+i+16*half
    v8bf a0 = *(const v8bf*)(arow + k0 + 16 * half);
    v8bf a1 = *(const v8bf*)(arow + k0 + 16 * half + 8);
    v16bf a = __builtin_shufflevector(a0, a1, 0, 1, 2, 3, 4, 5, 6, 7,
                                      8, 9, 10, 11, 12, 13, 14, 15);
    // B 32x16 fragment: element i -> K=k0+i+16*half (contiguous in Bp)
    v16bf b = *(const v16bf*)(Bp + ((size_t)(k0 >> 5) * Nc + ncol) * 32 + 16 * half);
    acc = __builtin_amdgcn_wmma_f32_16x16x32_bf16(false, a, false, b,
                                                  (short)0, acc, false, false);
  }
  // D layout: n = lq, m = r + 8*half
#pragma unroll
  for (int r = 0; r < 8; ++r)
    C[(size_t)(tm * 16 + r + 8 * half) * Nc + (tn * 16 + lq)] = acc[r];
}

template <int K>
static void gemm_t_launch(const __bf16* A, const __bf16* Bp, float* C,
                          int M, int Nc, hipStream_t s) {
  int nTiles = (M >> 4) * (Nc >> 4);
  int blocks = cdiv((long long)nTiles * 32, 256);
  wmma_gemm_t<K><<<blocks, 256, 0, s>>>(A, Bp, C, M, Nc);
}

// ---- per-node attention scores: s = sum_c h[n,hd,c] * a[hd,c] --------------
__global__ void scores_k(const float* __restrict__ h, const float* __restrict__ a_s,
                         const float* __restrict__ a_d, float* __restrict__ s_src,
                         float* __restrict__ s_dst, int n, int heads, int C) {
  int t = blockIdx.x * blockDim.x + threadIdx.x;
  if (t >= n * heads) return;
  int node = t / heads, hd = t % heads;
  const float* hp = h + (size_t)node * heads * C + (size_t)hd * C;
  float ss = 0.f, sd = 0.f;
  for (int c = 0; c < C; ++c) { float v = hp[c]; ss += v * a_s[hd * C + c]; sd += v * a_d[hd * C + c]; }
  s_src[t] = ss; s_dst[t] = sd;
}

__global__ void zero_k(float* p, long long n) {
  long long i = (long long)blockIdx.x * blockDim.x + threadIdx.x;
  if (i < n) p[i] = 0.f;
}

// ---- edge pass 1: segment max of leaky_relu scores over dst ----------------
__global__ void edge_max_k(const int* __restrict__ src, const int* __restrict__ dst,
                           const float* __restrict__ s_src, const float* __restrict__ s_dst,
                           unsigned* __restrict__ mord, int E, int heads) {
  int t = blockIdx.x * blockDim.x + threadIdx.x;
  if (t >= E * heads) return;
  int e = t / heads, hd = t % heads;
  float v = lrelu(s_src[src[e] * heads + hd] + s_dst[dst[e] * heads + hd]);
  atomicMax(&mord[dst[e] * heads + hd], f2ord(v));
}

// ---- edge pass 2: p = exp(e - m[dst]); z[dst] += p; cache p ----------------
__global__ void edge_sum_k(const int* __restrict__ src, const int* __restrict__ dst,
                           const float* __restrict__ s_src, const float* __restrict__ s_dst,
                           const unsigned* __restrict__ mord, float* __restrict__ z,
                           float* __restrict__ parr, int E, int heads) {
  int t = blockIdx.x * blockDim.x + threadIdx.x;
  if (t >= E * heads) return;
  int e = t / heads, hd = t % heads;
  int d = dst[e] * heads + hd;
  float v = lrelu(s_src[src[e] * heads + hd] + s_dst[d]);
  float p = expf(v - ord2f(mord[d]));
  parr[t] = p;
  atomicAdd(&z[d], p);
}

// ---- edge pass 3: out[dst,ch] += (p/z) * h[src,ch] -------------------------
__global__ void edge_aggr_k(const int* __restrict__ src, const int* __restrict__ dst,
                            const float* __restrict__ parr, const float* __restrict__ z,
                            const float* __restrict__ hfeat, float* __restrict__ out,
                            int E, int heads, int C) {
  long long t = (long long)blockIdx.x * blockDim.x + threadIdx.x;
  const int tc = heads * C;
  if (t >= (long long)E * tc) return;
  int e  = (int)(t / tc);
  int ch = (int)(t % tc);
  int hd = ch / C;
  int s = src[e], d = dst[e];
  float alpha = parr[(size_t)e * heads + hd] / (z[d * heads + hd] + 1e-16f);
  atomicAdd(&out[(size_t)d * tc + ch], alpha * hfeat[(size_t)s * tc + ch]);
}

// ---- x = prelu(in + b, p) --------------------------------------------------
__global__ void bias_prelu_k(const float* __restrict__ in, const float* __restrict__ b,
                             const float* __restrict__ p, float* __restrict__ out,
                             long long n, int tc) {
  long long i = (long long)blockIdx.x * blockDim.x + threadIdx.x;
  if (i >= n) return;
  int c = (int)(i % tc);
  float v = in[i] + b[c];
  out[i] = v > 0.f ? v : p[c] * v;
}

// ---- row-wise l2 normalize -------------------------------------------------
__global__ void l2norm_k(const float* __restrict__ in, float* __restrict__ out, int n, int c) {
  int i = blockIdx.x * blockDim.x + threadIdx.x;
  if (i >= n) return;
  const float* r = in + (size_t)i * c;
  float s = 0.f;
  for (int j = 0; j < c; ++j) s += r[j] * r[j];
  float inv = 1.f / (sqrtf(s) + 1e-12f);
  for (int j = 0; j < c; ++j) out[(size_t)i * c + j] = r[j] * inv;
}

// ---- gather [e1 | e2] concat rows, emit bf16 for the fusion GEMM -----------
__global__ void gather_cat_k(const float* __restrict__ x2, const int* __restrict__ idx,
                             __bf16* __restrict__ cat, int nb, int c) {
  int t = blockIdx.x * blockDim.x + threadIdx.x;
  int tot = nb * 2 * c;
  if (t >= tot) return;
  int i = t / (2 * c), ch = t % (2 * c);
  int node = (ch < c) ? idx[i] : idx[nb + i];
  int cc = (ch < c) ? ch : ch - c;
  cat[t] = f2bf(x2[(size_t)node * c + cc]);
}

// ---- decoder heads: log = relu(hpre+bf)@Wlog + blog (x2) -------------------
__global__ void decoder_k(const float* __restrict__ hpre, const float* __restrict__ bf,
                          const float* __restrict__ wlog, const float* __restrict__ blog,
                          const float* __restrict__ wlog1, const float* __restrict__ blog1,
                          float* __restrict__ out0, float* __restrict__ out1, int nb, int c) {
  int i = blockIdx.x * blockDim.x + threadIdx.x;
  if (i >= nb) return;
  float a0 = 0.f, a1 = 0.f;
  for (int j = 0; j < c; ++j) {
    float hv = hpre[(size_t)i * c + j] + bf[j];
    hv = hv > 0.f ? hv : 0.f;
    a0 += hv * wlog[j]; a1 += hv * wlog1[j];
  }
  out0[i] = a0 + blog[0];
  out1[i] = a1 + blog1[0];
}

// ---- adversary: precompute row-sums of adv_w and sum(adv_b) ----------------
__global__ void adv_prep_k(const float* __restrict__ advw, const float* __restrict__ advb,
                           float* __restrict__ wrs, int c) {
  int i = threadIdx.x;
  if (i < c)  { float s = 0.f; for (int j = 0; j < c; ++j) s += advw[i * c + j]; wrs[i] = s; }
  if (i == c) { float s = 0.f; for (int j = 0; j < c; ++j) s += advb[j];         wrs[c] = s; }
}
__global__ void adv_score_k(const float* __restrict__ x2, const float* __restrict__ wrs,
                            float* __restrict__ out, int n, int c) {
  int i = blockIdx.x * blockDim.x + threadIdx.x;
  if (i >= n) return;
  float s = wrs[c];
  for (int j = 0; j < c; ++j) s += x2[(size_t)i * c + j] * wrs[j];
  out[i] = s;
}

// ---------------------------------------------------------------------------
// host-side sequencing
// ---------------------------------------------------------------------------
static void run_encode(const float* x, const __bf16* W1p, const float* as1, const float* ad1,
                       const float* b1, const float* p1, const __bf16* W2p, const float* as2,
                       const float* ad2, const float* b2, const float* p2,
                       const int* src, const int* dst, float* ws, float* x2_out,
                       hipStream_t s) {
  float*    h1    = ws + OFF_H1;
  float*    aggr1 = ws + OFF_AGGR1;
  float*    ssrc  = ws + OFF_SSRC;
  float*    sdst  = ws + OFF_SDST;
  unsigned* mord  = (unsigned*)(ws + OFF_MORD);
  float*    z     = ws + OFF_Z;
  float*    parr  = ws + OFF_PARR;
  float*    aggr2 = ws + OFF_AGGR2;
  __bf16*   abf   = (__bf16*)(ws + OFF_ABF);

  // ---- layer 1: h1 = x @ W1  [N,128]@[128,256]
  conv_bf16_k<<<cdiv((long long)kN * kIn, 256), 256, 0, s>>>(x, abf, (long long)kN * kIn);
  gemm_t_launch<kIn>(abf, W1p, h1, kN, kTC1, s);
  scores_k<<<cdiv((long long)kN * kHeads, 256), 256, 0, s>>>(h1, as1, ad1, ssrc, sdst, kN, kHeads, kC1);
  zero_k<<<cdiv((long long)kN * kHeads, 256), 256, 0, s>>>((float*)mord, (long long)kN * kHeads);
  zero_k<<<cdiv((long long)kN * kHeads, 256), 256, 0, s>>>(z, (long long)kN * kHeads);
  zero_k<<<cdiv((long long)kN * kTC1, 256), 256, 0, s>>>(aggr1, (long long)kN * kTC1);
  edge_max_k<<<cdiv((long long)kE * kHeads, 256), 256, 0, s>>>(src, dst, ssrc, sdst, mord, kE, kHeads);
  edge_sum_k<<<cdiv((long long)kE * kHeads, 256), 256, 0, s>>>(src, dst, ssrc, sdst, mord, z, parr, kE, kHeads);
  edge_aggr_k<<<cdiv((long long)kE * kTC1, 256), 256, 0, s>>>(src, dst, parr, z, h1, aggr1, kE, kHeads, kC1);
  bias_prelu_k<<<cdiv((long long)kN * kTC1, 256), 256, 0, s>>>(aggr1, b1, p1, aggr1, (long long)kN * kTC1, kTC1);

  // ---- layer 2: h2 = x1 @ W2  [N,256]@[256,32]  (h2 reuses h1 region)
  conv_bf16_k<<<cdiv((long long)kN * kTC1, 256), 256, 0, s>>>(aggr1, abf, (long long)kN * kTC1);
  gemm_t_launch<kTC1>(abf, W2p, h1, kN, kC2, s);
  scores_k<<<cdiv((long long)kN, 256), 256, 0, s>>>(h1, as2, ad2, ssrc, sdst, kN, 1, kC2);
  zero_k<<<cdiv((long long)kN, 256), 256, 0, s>>>((float*)mord, (long long)kN);
  zero_k<<<cdiv((long long)kN, 256), 256, 0, s>>>(z, (long long)kN);
  zero_k<<<cdiv((long long)kN * kC2, 256), 256, 0, s>>>(aggr2, (long long)kN * kC2);
  edge_max_k<<<cdiv((long long)kE, 256), 256, 0, s>>>(src, dst, ssrc, sdst, mord, kE, 1);
  edge_sum_k<<<cdiv((long long)kE, 256), 256, 0, s>>>(src, dst, ssrc, sdst, mord, z, parr, kE, 1);
  edge_aggr_k<<<cdiv((long long)kE * kC2, 256), 256, 0, s>>>(src, dst, parr, z, h1, aggr2, kE, 1, kC2);
  bias_prelu_k<<<cdiv((long long)kN * kC2, 256), 256, 0, s>>>(aggr2, b2, p2, x2_out, (long long)kN * kC2, kC2);
}

extern "C" void kernel_launch(void* const* d_in, const int* in_sizes, int n_in,
                              void* d_out, int out_size, void* d_ws, size_t ws_size,
                              hipStream_t stream) {
  const float* x_o   = (const float*)d_in[0];
  const float* x_a   = (const float*)d_in[1];
  const float* W1    = (const float*)d_in[2];
  const float* as1   = (const float*)d_in[3];
  const float* ad1   = (const float*)d_in[4];
  const float* b1    = (const float*)d_in[5];
  const float* p1    = (const float*)d_in[6];
  const float* W2    = (const float*)d_in[7];
  const float* as2   = (const float*)d_in[8];
  const float* ad2   = (const float*)d_in[9];
  const float* b2    = (const float*)d_in[10];
  const float* p2    = (const float*)d_in[11];
  const float* advw  = (const float*)d_in[12];
  const float* advb  = (const float*)d_in[13];
  const float* Wq    = (const float*)d_in[14];
  const float* Wk    = (const float*)d_in[15];
  const float* Wf    = (const float*)d_in[16];
  const float* bfv   = (const float*)d_in[17];
  const float* Wlog  = (const float*)d_in[18];
  const float* blog  = (const float*)d_in[19];
  const float* Wlog1 = (const float*)d_in[20];
  const float* blog1 = (const float*)d_in[21];
  const int*   eidx  = (const int*)d_in[22];
  const int*   idx   = (const int*)d_in[23];
  const int* src = eidx;
  const int* dst = eidx + kE;

  float* out = (float*)d_out;
  float* ws  = (float*)d_ws;

  // output tuple layout (flat, reference return order)
  float* out_log    = out;                         // [4096]
  float* out_q      = out_log + kNB;               // [N,32]
  float* out_k      = out_q + (size_t)kN * kProj;  // [N,32]
  float* out_x2o    = out_k + (size_t)kN * kProj;  // [N,32]
  float* out_logits = out_x2o + (size_t)kN * kC2;  // [1, 2N]
  float* out_log1   = out_logits + 2 * (size_t)kN; // [4096]

  float*  x2a   = ws + OFF_X2A;
  float*  hdec  = ws + OFF_HDEC;
  float*  wrs   = ws + OFF_WRS;
  float*  tmp   = ws + OFF_H1;                      // scratch for q/k raw proj
  __bf16* x2obf = (__bf16*)(ws + OFF_X2OBF);
  __bf16* x2abf = (__bf16*)(ws + OFF_X2ABF);
  __bf16* catbf = (__bf16*)(ws + OFF_CATBF);
  __bf16* W1p   = (__bf16*)(ws + OFF_W1P);
  __bf16* W2p   = (__bf16*)(ws + OFF_W2P);
  __bf16* Wqp   = (__bf16*)(ws + OFF_WQP);
  __bf16* Wkp   = (__bf16*)(ws + OFF_WKP);
  __bf16* Wfp   = (__bf16*)(ws + OFF_WFP);

  // pack all GEMM weights into bf16 fragment layout (tiny, per-call, deterministic)
  pack_b_k<<<cdiv(kIn * kTC1, 256), 256, 0, stream>>>(W1, W1p, kIn, kTC1);
  pack_b_k<<<cdiv(kTC1 * kC2, 256), 256, 0, stream>>>(W2, W2p, kTC1, kC2);
  pack_b_k<<<cdiv(kC2 * kProj, 256), 256, 0, stream>>>(Wq, Wqp, kC2, kProj);
  pack_b_k<<<cdiv(kC2 * kProj, 256), 256, 0, stream>>>(Wk, Wkp, kC2, kProj);
  pack_b_k<<<cdiv(2 * kC2 * kC2, 256), 256, 0, stream>>>(Wf, Wfp, 2 * kC2, kC2);

  // two encoder passes
  run_encode(x_o, W1p, as1, ad1, b1, p1, W2p, as2, ad2, b2, p2, src, dst, ws, out_x2o, stream);
  run_encode(x_a, W1p, as1, ad1, b1, p1, W2p, as2, ad2, b2, p2, src, dst, ws, x2a, stream);

  // bf16 copies of encoder outputs for the small projections
  conv_bf16_k<<<cdiv((long long)kN * kC2, 256), 256, 0, stream>>>(out_x2o, x2obf, (long long)kN * kC2);
  conv_bf16_k<<<cdiv((long long)kN * kC2, 256), 256, 0, stream>>>(x2a, x2abf, (long long)kN * kC2);

  // MoCo projections, l2-normalized
  gemm_t_launch<kC2>(x2obf, Wqp, tmp, kN, kProj, stream);
  l2norm_k<<<cdiv(kN, 256), 256, 0, stream>>>(tmp, out_q, kN, kProj);
  gemm_t_launch<kC2>(x2abf, Wkp, tmp, kN, kProj, stream);
  l2norm_k<<<cdiv(kN, 256), 256, 0, stream>>>(tmp, out_k, kN, kProj);

  // fusion decoder
  gather_cat_k<<<cdiv((long long)kNB * 2 * kC2, 256), 256, 0, stream>>>(out_x2o, idx, catbf, kNB, kC2);
  gemm_t_launch<2 * kC2>(catbf, Wfp, hdec, kNB, kC2, stream);
  decoder_k<<<cdiv(kNB, 256), 256, 0, stream>>>(hdec, bfv, Wlog, blog, Wlog1, blog1,
                                                out_log, out_log1, kNB, kC2);

  // adversarial discriminator logits
  adv_prep_k<<<1, 64, 0, stream>>>(advw, advb, wrs, kC2);
  adv_score_k<<<cdiv(kN, 256), 256, 0, stream>>>(out_x2o, wrs, out_logits, kN, kC2);
  adv_score_k<<<cdiv(kN, 256), 256, 0, stream>>>(x2a, wrs, out_logits + kN, kN, kC2);
}
